// LocMatcher_64845416235456
// MI455X (gfx1250) — matrix operand order repaired and checked by reading, not compiled, not run on hardware
//
#include <hip/hip_runtime.h>
#include <stdint.h>

#define BATCH 32
#define SEQ   1024
#define DIM   512
#define NH    8
#define DHD   64
#define NL    4
#define TOK   (BATCH * SEQ)   /* 32768 */
#define QKVN  (3 * DIM)       /* 1536  */
#define QKN   1024            /* q+k row width in qkv buffer */

typedef __attribute__((ext_vector_type(16))) __bf16 v16bf;
typedef __attribute__((ext_vector_type(8)))  float  v8f;
typedef __attribute__((ext_vector_type(4)))  unsigned int v4u;
typedef __attribute__((ext_vector_type(8)))  int    v8i;
typedef __attribute__((ext_vector_type(4)))  int    v4i;
typedef __attribute__((address_space(3)))    unsigned short lds_us_t;

union AFrag { v16bf v; unsigned short u[16]; };

#if __has_builtin(__builtin_amdgcn_tensor_load_to_lds) && \
    __has_builtin(__builtin_amdgcn_s_wait_tensorcnt)
#define USE_TDM 1
#else
#define USE_TDM 0
#endif

__device__ __forceinline__ unsigned short f2b(float f) {
  union { float f; unsigned int u; } c; c.f = f;
  unsigned int u = c.u;
  u += 0x7fffu + ((u >> 16) & 1u);          // round-to-nearest-even
  return (unsigned short)(u >> 16);
}

#if USE_TDM
__device__ __forceinline__ unsigned lds_byte_offset(const unsigned short* p) {
  return (unsigned)(size_t)(lds_us_t*)p;
}

// D# group1 for a 2D bf16 tile load with LDS row padding.
// pad_interval code c -> pad after 2^(c+1) DWORDs of data;
// pad_amount code a -> (a+1) DWORDs of pad inserted.
__device__ __forceinline__ v8i tdm_desc_g1_2d(unsigned tile_w, unsigned tile_h,
                                              unsigned stride0,
                                              unsigned pad_int, unsigned pad_amt) {
  v8i g = {0, 0, 0, 0, 0, 0, 0, 0};
  g[0] = (int)((1u << 16) | (1u << 20) | (pad_int << 22) | (pad_amt << 25));
  g[1] = (int)((tile_w & 0xFFFFu) << 16);                      // tensor_dim0 lo
  g[2] = (int)((tile_w >> 16) | ((tile_h & 0xFFFFu) << 16));   // dim0 hi|dim1 lo
  g[3] = (int)((tile_h >> 16) | (tile_w << 16));               // dim1 hi|tile_dim0
  g[4] = (int)(tile_h & 0xFFFFu);                              // tile_dim1, dim2=0
  g[5] = (int)stride0;                                         // dim0_stride lo32
  return g;
}

__device__ __forceinline__ void tdm_load_2d(const unsigned short* gptr,
                                            unsigned lds_off, v8i g1) {
  unsigned long long ga = (unsigned long long)(size_t)gptr;
  v4u g0;
  g0[0] = 1u;                                   // count=1, user descriptor
  g0[1] = lds_off;                              // LDS byte address
  g0[2] = (unsigned)ga;                         // global addr lo
  g0[3] = (unsigned)((ga >> 32) & 0x01FFFFFFu) | (2u << 30);  // addr hi | type=2
  v4i z4 = {0, 0, 0, 0};
#if __clang_major__ >= 23
  v8i z8 = {0, 0, 0, 0, 0, 0, 0, 0};
  __builtin_amdgcn_tensor_load_to_lds(g0, g1, z4, z4, z8, 0);
#else
  __builtin_amdgcn_tensor_load_to_lds(g0, g1, z4, z4, 0);
#endif
}
#endif  // USE_TDM

// ---------------------------------------------------------------------------
// bf16 WMMA GEMM: C[M,N] = A[M,K] * BT[N,K]^T + bias, optional ReLU.
// Tile 128x128x32, 128 threads (4 waves); each wave: 2 A frags x 8 B frags =
// 16 WMMAs per K-chunk. With TDM: double-buffered LDS tiles filled by the
// Tensor Data Mover (wave 0 issues, s_wait_tensorcnt retires) overlapping DMA
// with WMMA compute. Fallback: staged b128 copies.
// ---------------------------------------------------------------------------
__global__ __launch_bounds__(128)
void gemm_bf16(const unsigned short* __restrict__ A,    // [M][K]
               const unsigned short* __restrict__ BT,   // [N][K] pre-transposed
               const float* __restrict__ bias,
               float* __restrict__ Cf,
               unsigned short* __restrict__ Cb,
               unsigned short* __restrict__ Vtg,
               int M, int N, int K, int relu, int qkvmode)
{
#if USE_TDM
  __shared__ unsigned short As[2][128][40];
  __shared__ unsigned short Bs[2][128][40];
#else
  __shared__ unsigned short As[1][128][40];
  __shared__ unsigned short Bs[1][128][40];
#endif

  const int tid  = threadIdx.x;
  const int wave = tid >> 5;
  const int lane = tid & 31;
  const int half = lane >> 4;
  const int l15  = lane & 15;
  const int m0   = blockIdx.y << 7;
  const int n0   = blockIdx.x << 7;

  v8f acc[2][8];
  #pragma unroll
  for (int t = 0; t < 2; ++t)
    #pragma unroll
    for (int j = 0; j < 8; ++j)
      #pragma unroll
      for (int e = 0; e < 8; ++e) acc[t][j][e] = 0.0f;

  auto do_chunk = [&](const unsigned short (*Asb)[40],
                      const unsigned short (*Bsb)[40]) {
    AFrag a[2];
    #pragma unroll
    for (int t = 0; t < 2; ++t) {
      const unsigned short* ap = &Asb[(wave << 5) + (t << 4) + l15][half << 4];
      #pragma unroll
      for (int i = 0; i < 8; ++i) { a[t].u[i] = ap[i]; a[t].u[8 + i] = ap[8 + i]; }
    }
    #pragma unroll
    for (int j = 0; j < 8; ++j) {
      AFrag bf;
      const unsigned short* bp = &Bsb[(j << 4) + l15][half << 4];
      #pragma unroll
      for (int i = 0; i < 16; ++i) bf.u[i] = bp[i];
      #pragma unroll
      for (int t = 0; t < 2; ++t)
        acc[t][j] = __builtin_amdgcn_wmma_f32_16x16x32_bf16(
            false, a[t].v, false, bf.v, (short)0, acc[t][j], false, false);
    }
  };

#if USE_TDM
  const unsigned asOff = lds_byte_offset(&As[0][0][0]);
  const unsigned bsOff = lds_byte_offset(&Bs[0][0][0]);
  const unsigned BUFB  = 128 * 40 * 2;          // bytes per LDS buffer
  const v8i g1 = tdm_desc_g1_2d(32u, 128u, (unsigned)K, 3u, 3u);  // 64B+16B pad
  const int nk = K >> 5;
  if (wave == 0) {
    tdm_load_2d(A + (size_t)m0 * K, asOff, g1);
    tdm_load_2d(BT + (size_t)n0 * K, bsOff, g1);
  }
  for (int it = 0; it < nk; ++it) {
    const int cur = it & 1;
    if (wave == 0) {
      if (it + 1 < nk) {
        const int k1 = (it + 1) << 5;
        tdm_load_2d(A + (size_t)m0 * K + k1, asOff + (unsigned)(cur ^ 1) * BUFB, g1);
        tdm_load_2d(BT + (size_t)n0 * K + k1, bsOff + (unsigned)(cur ^ 1) * BUFB, g1);
        __builtin_amdgcn_s_wait_tensorcnt((short)2);   // current buffer complete
      } else {
        __builtin_amdgcn_s_wait_tensorcnt((short)0);
      }
    }
    __syncthreads();            // current buffer visible to all waves
    do_chunk(As[cur], Bs[cur]);
    __syncthreads();            // all reads done before next DMA overwrites
  }
#else
  for (int k0 = 0; k0 < K; k0 += 32) {
    __syncthreads();
    uint4 ga[4], gb[4];
    #pragma unroll
    for (int q = 0; q < 4; ++q)
      ga[q] = *(const uint4*)&A[(size_t)(m0 + tid) * K + k0 + (q << 3)];
    #pragma unroll
    for (int q = 0; q < 4; ++q) {
      int c = tid + (q << 7);
      gb[q] = *(const uint4*)&BT[(size_t)(n0 + (c >> 2)) * K + k0 + ((c & 3) << 3)];
    }
    #pragma unroll
    for (int q = 0; q < 4; ++q)
      *(uint4*)&As[0][tid][q << 3] = ga[q];
    #pragma unroll
    for (int q = 0; q < 4; ++q) {
      int c = tid + (q << 7);
      *(uint4*)&Bs[0][c >> 2][(c & 3) << 3] = gb[q];
    }
    __syncthreads();
    do_chunk(As[0], Bs[0]);
  }
#endif

  // epilogue: C/D layout row = half*8+r, col = lane&15
  #pragma unroll
  for (int t = 0; t < 2; ++t)
    #pragma unroll
    for (int j = 0; j < 8; ++j) {
      int col = n0 + (j << 4) + l15;
      float bv = bias ? bias[col] : 0.0f;
      #pragma unroll
      for (int r = 0; r < 8; ++r) {
        int row = m0 + (wave << 5) + (t << 4) + (half << 3) + r;
        float v = acc[t][j][r] + bv;
        if (relu) v = fmaxf(v, 0.0f);
        if (qkvmode) {
          if (col < QKN) {
            Cb[(size_t)row * QKN + col] = f2b(v);
          } else {
            int hh = (col - QKN) >> 6, dd = (col - QKN) & 63;
            int bb = row >> 10, ss = row & (SEQ - 1);
            Vtg[(((size_t)bb * NH + hh) * DHD + dd) * SEQ + ss] = f2b(v);
          }
        } else {
          size_t idx = (size_t)row * N + col;
          if (Cf) Cf[idx] = v;
          if (Cb) Cb[idx] = f2b(v);
        }
      }
    }
}

// ---------------------------------------------------------------------------
// Flash attention. Block = (b, h, 128-q-row slice); 8 waves, each owns a
// 16-row q tile with online softmax. With TDM: double-buffered K/V chunk
// staging via the Tensor Data Mover. K tile [key][dim] (144B LDS stride),
// V tile [dim][key] from pre-transposed buffer (272B LDS stride).
// ---------------------------------------------------------------------------
__global__ __launch_bounds__(256)
void attn_flash(const unsigned short* __restrict__ qk,   // [tok][1024] q|k
                const unsigned short* __restrict__ vt,   // [b][h][d][s]
                const int* __restrict__ dlen,
                unsigned short* __restrict__ ctx)
{
#if USE_TDM
  __shared__ unsigned short Ks[2][128][72];
  __shared__ unsigned short Vt[2][64][136];
#else
  __shared__ unsigned short Ks[1][128][72];
  __shared__ unsigned short Vt[1][64][136];
#endif
  __shared__ unsigned short Ps[8][16][32];   // per-wave C->A layout bounce

  const int tid  = threadIdx.x;
  const int wave = tid >> 5;
  const int lane = tid & 31;
  const int half = lane >> 4;
  const int l15  = lane & 15;

  const int qc = blockIdx.x & 7;
  const int bh = blockIdx.x >> 3;
  const int h  = bh & (NH - 1);
  const int b  = bh >> 3;
  const int q0 = (qc << 7) + (wave << 4);
  const int len = dlen[b];
  const float scale = 0.125f;   // 1/sqrt(64)
  const unsigned short* vhead = vt + (size_t)bh * DHD * SEQ;

  // q fragments: A layout, two K=32 chunks over head dims 0..63
  AFrag qa0, qa1;
  {
    const unsigned short* qp = qk + (size_t)(b * SEQ + q0 + l15) * QKN + h * DHD;
    #pragma unroll
    for (int i = 0; i < 8; ++i) {
      qa0.u[i]     = qp[(half << 4) + i];
      qa0.u[8 + i] = qp[8 + (half << 4) + i];
      qa1.u[i]     = qp[32 + (half << 4) + i];
      qa1.u[8 + i] = qp[40 + (half << 4) + i];
    }
  }

  float mrow[8], lrow[8];
  v8f oacc[4];
  #pragma unroll
  for (int r = 0; r < 8; ++r) { mrow[r] = -3.0e38f; lrow[r] = 0.0f; }
  #pragma unroll
  for (int j = 0; j < 4; ++j)
    #pragma unroll
    for (int e = 0; e < 8; ++e) oacc[j][e] = 0.0f;

  // one 32-key step group against the staged chunk starting at kc
  auto attn_chunk = [&](int kc, const unsigned short (*Ksb)[72],
                        const unsigned short (*Vtb)[136]) {
    for (int kb = 0; kb < 128; kb += 32) {
      if (kc + kb >= len) break;             // uniform
      v8f s0, s1;
      #pragma unroll
      for (int e = 0; e < 8; ++e) { s0[e] = 0.0f; s1[e] = 0.0f; }

      AFrag kf;
      #pragma unroll
      for (int i = 0; i < 16; ++i) kf.u[i] = Ksb[kb + l15][(half << 4) + i];
      s0 = __builtin_amdgcn_wmma_f32_16x16x32_bf16(false, qa0.v, false, kf.v,
                                                   (short)0, s0, false, false);
      #pragma unroll
      for (int i = 0; i < 16; ++i) kf.u[i] = Ksb[kb + l15][32 + (half << 4) + i];
      s0 = __builtin_amdgcn_wmma_f32_16x16x32_bf16(false, qa1.v, false, kf.v,
                                                   (short)0, s0, false, false);
      #pragma unroll
      for (int i = 0; i < 16; ++i) kf.u[i] = Ksb[kb + 16 + l15][(half << 4) + i];
      s1 = __builtin_amdgcn_wmma_f32_16x16x32_bf16(false, qa0.v, false, kf.v,
                                                   (short)0, s1, false, false);
      #pragma unroll
      for (int i = 0; i < 16; ++i) kf.u[i] = Ksb[kb + 16 + l15][32 + (half << 4) + i];
      s1 = __builtin_amdgcn_wmma_f32_16x16x32_bf16(false, qa1.v, false, kf.v,
                                                   (short)0, s1, false, false);

      const bool ok0 = (kc + kb + l15) < len;
      const bool ok1 = (kc + kb + 16 + l15) < len;
      float p0[8], p1[8], corr[8];
      #pragma unroll
      for (int r = 0; r < 8; ++r) {
        float a0 = ok0 ? s0[r] * scale : -1e9f;
        float a1 = ok1 ? s1[r] * scale : -1e9f;
        float mx = fmaxf(a0, a1);
        #pragma unroll
        for (int d = 1; d < 16; d <<= 1) mx = fmaxf(mx, __shfl_xor(mx, d, 32));
        float mnew = fmaxf(mrow[r], mx);
        float cf = __expf(mrow[r] - mnew);
        float e0 = __expf(a0 - mnew);
        float e1 = __expf(a1 - mnew);
        float sum = e0 + e1;
        #pragma unroll
        for (int d = 1; d < 16; d <<= 1) sum += __shfl_xor(sum, d, 32);
        lrow[r] = lrow[r] * cf + sum;
        mrow[r] = mnew;
        corr[r] = cf;
        p0[r] = e0;
        p1[r] = e1;
      }
      #pragma unroll
      for (int j = 0; j < 4; ++j)
        #pragma unroll
        for (int r = 0; r < 8; ++r) oacc[j][r] *= corr[r];

      // C layout -> A layout bounce through per-wave LDS tile
      #pragma unroll
      for (int r = 0; r < 8; ++r) {
        Ps[wave][(half << 3) + r][l15]      = f2b(p0[r]);
        Ps[wave][(half << 3) + r][16 + l15] = f2b(p1[r]);
      }
      AFrag pa;
      {
        const unsigned short* pp = &Ps[wave][l15][half << 4];
        #pragma unroll
        for (int i = 0; i < 8; ++i) { pa.u[i] = pp[i]; pa.u[8 + i] = pp[8 + i]; }
      }
      #pragma unroll
      for (int j = 0; j < 4; ++j) {
        AFrag vf;                  // from transposed V: contiguous per lane
        const unsigned short* vp = &Vtb[(j << 4) + l15][kb + (half << 4)];
        #pragma unroll
        for (int i = 0; i < 16; ++i) vf.u[i] = vp[i];
        oacc[j] = __builtin_amdgcn_wmma_f32_16x16x32_bf16(
            false, pa.v, false, vf.v, (short)0, oacc[j], false, false);
      }
    }
  };

#if USE_TDM
  const unsigned ksOff = lds_byte_offset(&Ks[0][0][0]);
  const unsigned vtOff = lds_byte_offset(&Vt[0][0][0]);
  const unsigned KSB = 128 * 72 * 2, VTB = 64 * 136 * 2;
  // K: 128B data + 16B pad per row (codes 4,3); V: 256B data + 16B pad (5,3)
  const v8i gK = tdm_desc_g1_2d(64u, 128u, QKN, 4u, 3u);
  const v8i gV = tdm_desc_g1_2d(128u, 64u, SEQ, 5u, 3u);
  const unsigned short* kbase = qk + (size_t)b * SEQ * QKN + DIM + h * DHD;
  const int lenc = len < SEQ ? len : SEQ;
  const int nc = (lenc + 127) >> 7;
  if (wave == 0) {
    tdm_load_2d(kbase, ksOff, gK);
    tdm_load_2d(vhead, vtOff, gV);
  }
  for (int ic = 0; ic < nc; ++ic) {
    const int cur = ic & 1;
    const int kc = ic << 7;
    if (wave == 0) {
      if (ic + 1 < nc) {
        tdm_load_2d(kbase + (size_t)(kc + 128) * QKN, ksOff + (unsigned)(cur ^ 1) * KSB, gK);
        tdm_load_2d(vhead + kc + 128, vtOff + (unsigned)(cur ^ 1) * VTB, gV);
        __builtin_amdgcn_s_wait_tensorcnt((short)2);
      } else {
        __builtin_amdgcn_s_wait_tensorcnt((short)0);
      }
    }
    __syncthreads();
    attn_chunk(kc, Ks[cur], Vt[cur]);
    __syncthreads();
  }
#else
  for (int kc = 0; kc < SEQ; kc += 128) {
    if (kc >= len) break;                    // uniform: len is per-block
    __syncthreads();
    uint4 gk[4], gv[4];
    #pragma unroll
    for (int q = 0; q < 4; ++q) {            // K: 128 keys x 64 dims
      int c = tid + (q << 8);
      int r = c >> 3, col = (c & 7) << 3;
      gk[q] = *(const uint4*)&qk[(size_t)(b * SEQ + kc + r) * QKN + DIM + h * DHD + col];
    }
    #pragma unroll
    for (int q = 0; q < 4; ++q) {            // V: 64 dims x 128 keys
      int c = tid + (q << 8);
      int d = c >> 4, col = (c & 15) << 3;
      gv[q] = *(const uint4*)&vhead[(size_t)d * SEQ + kc + col];
    }
    #pragma unroll
    for (int q = 0; q < 4; ++q) {
      int c = tid + (q << 8);
      *(uint4*)&Ks[0][c >> 3][(c & 7) << 3] = gk[q];
    }
    #pragma unroll
    for (int q = 0; q < 4; ++q) {
      int c = tid + (q << 8);
      *(uint4*)&Vt[0][c >> 4][(c & 15) << 3] = gv[q];
    }
    __syncthreads();
    attn_chunk(kc, Ks[0], Vt[0]);
  }
#endif

  #pragma unroll
  for (int j = 0; j < 4; ++j)
    #pragma unroll
    for (int r = 0; r < 8; ++r) {
      int row = q0 + (half << 3) + r;
      float v = oacc[j][r] / lrow[r];
      ctx[(size_t)(b * SEQ + row) * DIM + h * DHD + (j << 4) + l15] = f2b(v);
    }
}

// ---------------------------------------------------------------------------
// Embedding: x = concat(loc[16], time_dist@Wtd+btd [3]) @ Wemb + bemb
// ---------------------------------------------------------------------------
__global__ __launch_bounds__(256)
void embed_kernel(const float* __restrict__ loc, const float* __restrict__ tds,
                  const float* __restrict__ Wtd, const float* __restrict__ btd,
                  const float* __restrict__ Wemb, const float* __restrict__ bemb,
                  float* __restrict__ xf, unsigned short* __restrict__ xb)
{
  size_t idx = (size_t)blockIdx.x * 256 + threadIdx.x;
  int d = (int)(idx & (DIM - 1));
  size_t tok = idx >> 9;
  const float* lp = loc + tok * 16;
  const float* tp = tds + tok * 8;
  float acc = bemb[d];
  #pragma unroll
  for (int i = 0; i < 16; ++i) acc += lp[i] * Wemb[i * DIM + d];
  #pragma unroll
  for (int j = 0; j < 3; ++j) {
    float td = btd[j];
    #pragma unroll
    for (int i = 0; i < 8; ++i) td += tp[i] * Wtd[i * 3 + j];
    acc += td * Wemb[(16 + j) * DIM + d];
  }
  xf[idx] = acc;
  xb[idx] = f2b(acc);
}

// ---------------------------------------------------------------------------
// x = LayerNorm(x + y) * g + b ; writes both f32 and bf16 copies.
// ---------------------------------------------------------------------------
__global__ __launch_bounds__(256)
void residual_ln(const float* __restrict__ x, const float* __restrict__ y,
                 const float* __restrict__ g, const float* __restrict__ be,
                 float* __restrict__ xo, unsigned short* __restrict__ xob)
{
  __shared__ float red[256];
  const int t = blockIdx.x, tid = threadIdx.x;
  const size_t base = (size_t)t * DIM;
  float v0 = x[base + tid] + y[base + tid];
  float v1 = x[base + 256 + tid] + y[base + 256 + tid];

  red[tid] = v0 + v1; __syncthreads();
  for (int o = 128; o > 0; o >>= 1) {
    if (tid < o) red[tid] += red[tid + o];
    __syncthreads();
  }
  float mean = red[0] * (1.0f / DIM);
  __syncthreads();

  float d0 = v0 - mean, d1 = v1 - mean;
  red[tid] = d0 * d0 + d1 * d1; __syncthreads();
  for (int o = 128; o > 0; o >>= 1) {
    if (tid < o) red[tid] += red[tid + o];
    __syncthreads();
  }
  float rstd = rsqrtf(red[0] * (1.0f / DIM) + 1e-5f);

  float o0 = d0 * rstd * g[tid] + be[tid];
  float o1 = d1 * rstd * g[256 + tid] + be[256 + tid];
  xo[base + tid] = o0;         xob[base + tid] = f2b(o0);
  xo[base + 256 + tid] = o1;   xob[base + 256 + tid] = f2b(o1);
}

// ---------------------------------------------------------------------------
// Addr branch: a[b,:] = [addr_b, poi_emb[addr_type_b]] @ Wa + ba
// ---------------------------------------------------------------------------
__global__ __launch_bounds__(1024)
void addr_kernel(const float* __restrict__ addr, const float* __restrict__ poi,
                 const int* __restrict__ atype, const float* __restrict__ Wa,
                 const float* __restrict__ ba, float* __restrict__ aa)
{
  int t = threadIdx.x;
  if (t >= BATCH * 32) return;
  int b = t >> 5, j = t & 31;
  int pt = atype[b];
  float acc = ba[j]
            + addr[b]          * Wa[0 * 32 + j]
            + poi[pt * 3 + 0]  * Wa[1 * 32 + j]
            + poi[pt * 3 + 1]  * Wa[2 * 32 + j]
            + poi[pt * 3 + 2]  * Wa[3 * 32 + j];
  aa[b * 32 + j] = acc;
}

// ---------------------------------------------------------------------------
// o[token] = sum_j tanh(x[token]@Wout[:,j] + a[b,j]) * Wcomb[j]
// ---------------------------------------------------------------------------
__global__ __launch_bounds__(256)
void final_kernel(const float* __restrict__ xf, const float* __restrict__ Wout,
                  const float* __restrict__ aa, const float* __restrict__ Wcomb,
                  float* __restrict__ oo)
{
  int gw = (int)(((size_t)blockIdx.x * 256 + threadIdx.x) >> 5);
  int lane = threadIdx.x & 31;
  if (gw >= TOK) return;
  int b = gw >> 10;                         // SEQ = 1024
  const float* xp = xf + (size_t)gw * DIM;
  float acc = 0.0f;
  for (int i = 0; i < DIM; ++i) acc += xp[i] * Wout[i * 32 + lane];
  float t = tanhf(acc + aa[b * 32 + lane]) * Wcomb[lane];
  #pragma unroll
  for (int d = 1; d < 32; d <<= 1) t += __shfl_xor(t, d, 32);
  if (lane == 0) oo[gw] = t;
}

// ---------------------------------------------------------------------------
// Masked log-softmax over sequence per batch element.
// ---------------------------------------------------------------------------
__global__ __launch_bounds__(256)
void lsm_kernel(const float* __restrict__ o, const int* __restrict__ dlen,
                float* __restrict__ out)
{
  __shared__ float red[256];
  const int b = blockIdx.x, tid = threadIdx.x;
  const int len = dlen[b];
  float v[4], mx = -3.0e38f;
  #pragma unroll
  for (int q = 0; q < 4; ++q) {
    int s = q * 256 + tid;
    float lv = (s < len) ? o[b * SEQ + s] : -1e9f;
    v[q] = lv;
    mx = fmaxf(mx, lv);
  }
  red[tid] = mx; __syncthreads();
  for (int off = 128; off > 0; off >>= 1) {
    if (tid < off) red[tid] = fmaxf(red[tid], red[tid + off]);
    __syncthreads();
  }
  float M = red[0]; __syncthreads();
  float sum = 0.0f;
  #pragma unroll
  for (int q = 0; q < 4; ++q) sum += __expf(v[q] - M);
  red[tid] = sum; __syncthreads();
  for (int off = 128; off > 0; off >>= 1) {
    if (tid < off) red[tid] += red[tid + off];
    __syncthreads();
  }
  float lz = M + __logf(red[0]);
  #pragma unroll
  for (int q = 0; q < 4; ++q) {
    int s = q * 256 + tid;
    out[b * SEQ + s] = (s < len) ? (v[q] - lz) : 0.0f;
  }
}

// ---------------------------------------------------------------------------
// f32 -> bf16 with transpose: per layer (blockIdx.y), src [512][N] -> dst
// [N][512]. Shift-only indexing (K = 512 fixed) -- no integer division.
// ---------------------------------------------------------------------------
__global__ __launch_bounds__(256)
void cvt_transpose512(const float* __restrict__ src,
                      unsigned short* __restrict__ dst, int N)
{
  int i = blockIdx.x * 256 + threadIdx.x;
  if (i >= N * 512) return;
  int n = i >> 9, k = i & 511;
  size_t base = (size_t)blockIdx.y * N * 512;
  dst[base + (size_t)n * 512 + k] = f2b(src[base + (size_t)k * N + n]);
}

// ---------------------------------------------------------------------------
extern "C" void kernel_launch(void* const* d_in, const int* in_sizes, int n_in,
                              void* d_out, int out_size, void* d_ws, size_t ws_size,
                              hipStream_t stream)
{
  (void)in_sizes; (void)n_in; (void)out_size; (void)ws_size;

  const float* addr  = (const float*)d_in[0];
  const float* loc   = (const float*)d_in[1];
  const float* tds   = (const float*)d_in[2];
  const float* Wtd   = (const float*)d_in[3];
  const float* btd   = (const float*)d_in[4];
  const float* Wemb  = (const float*)d_in[5];
  const float* bemb  = (const float*)d_in[6];
  const float* Wqkv  = (const float*)d_in[7];
  const float* bqkv  = (const float*)d_in[8];
  const float* Wo    = (const float*)d_in[9];
  const float* bo    = (const float*)d_in[10];
  const float* ln1g  = (const float*)d_in[11];
  const float* ln1b  = (const float*)d_in[12];
  const float* W1    = (const float*)d_in[13];
  const float* b1    = (const float*)d_in[14];
  const float* W2    = (const float*)d_in[15];
  const float* b2    = (const float*)d_in[16];
  const float* ln2g  = (const float*)d_in[17];
  const float* ln2b  = (const float*)d_in[18];
  const float* poi   = (const float*)d_in[19];
  const float* Wa    = (const float*)d_in[20];
  const float* ba    = (const float*)d_in[21];
  const float* Wout  = (const float*)d_in[22];
  const float* Wcomb = (const float*)d_in[23];
  const int*   atype = (const int*)d_in[24];
  const int*   dlen  = (const int*)d_in[25];
  float* out = (float*)d_out;

  char* w = (char*)d_ws;
  auto carve = [&](size_t bytes) -> char* {
    char* p = w;
    w += (bytes + 255) & ~(size_t)255;
    return p;
  };
  float*          xf    = (float*)carve((size_t)TOK * DIM * 4);
  unsigned short* xb    = (unsigned short*)carve((size_t)TOK * DIM * 2);
  unsigned short* qkb   = (unsigned short*)carve((size_t)TOK * QKN * 2);  // q|k
  unsigned short* vtb   = (unsigned short*)carve((size_t)TOK * DIM * 2);  // [b][h][d][s]
  unsigned short* ctxb  = (unsigned short*)carve((size_t)TOK * DIM * 2);
  float*          yf    = (float*)carve((size_t)TOK * DIM * 4);
  unsigned short* hb    = (unsigned short*)carve((size_t)TOK * DIM * 2);
  unsigned short* wqkvT = (unsigned short*)carve((size_t)NL * DIM * QKVN * 2);
  unsigned short* woT   = (unsigned short*)carve((size_t)NL * DIM * DIM * 2);
  unsigned short* w1T   = (unsigned short*)carve((size_t)NL * DIM * DIM * 2);
  unsigned short* w2T   = (unsigned short*)carve((size_t)NL * DIM * DIM * 2);
  float*          aa    = (float*)carve((size_t)BATCH * 32 * 4);
  float*          oo    = (float*)carve((size_t)TOK * 4);

  // weight preparation: f32 -> bf16 with [K][N] -> [N][K] transpose per layer
  cvt_transpose512<<<dim3((QKVN * 512) / 256, NL), 256, 0, stream>>>(Wqkv, wqkvT, QKVN);
  cvt_transpose512<<<dim3((DIM * 512) / 256, NL), 256, 0, stream>>>(Wo, woT, DIM);
  cvt_transpose512<<<dim3((DIM * 512) / 256, NL), 256, 0, stream>>>(W1, w1T, DIM);
  cvt_transpose512<<<dim3((DIM * 512) / 256, NL), 256, 0, stream>>>(W2, w2T, DIM);

  embed_kernel<<<(TOK * DIM) / 256, 256, 0, stream>>>(loc, tds, Wtd, btd,
                                                      Wemb, bemb, xf, xb);

  for (int l = 0; l < NL; ++l) {
    gemm_bf16<<<dim3(QKVN / 128, TOK / 128), 128, 0, stream>>>(
        xb, wqkvT + (size_t)l * DIM * QKVN, bqkv + l * QKVN,
        nullptr, qkb, vtb, TOK, QKVN, DIM, 0, 1);

    attn_flash<<<BATCH * NH * 8, 256, 0, stream>>>(qkb, vtb, dlen, ctxb);

    gemm_bf16<<<dim3(DIM / 128, TOK / 128), 128, 0, stream>>>(
        ctxb, woT + (size_t)l * DIM * DIM, bo + l * DIM,
        yf, nullptr, nullptr, TOK, DIM, DIM, 0, 0);

    residual_ln<<<TOK, 256, 0, stream>>>(xf, yf, ln1g + l * DIM,
                                         ln1b + l * DIM, xf, xb);

    gemm_bf16<<<dim3(DIM / 128, TOK / 128), 128, 0, stream>>>(
        xb, w1T + (size_t)l * DIM * DIM, b1 + l * DIM,
        nullptr, hb, nullptr, TOK, DIM, DIM, 1, 0);

    gemm_bf16<<<dim3(DIM / 128, TOK / 128), 128, 0, stream>>>(
        hb, w2T + (size_t)l * DIM * DIM, b2 + l * DIM,
        yf, nullptr, nullptr, TOK, DIM, DIM, 0, 0);

    residual_ln<<<TOK, 256, 0, stream>>>(xf, yf, ln2g + l * DIM,
                                         ln2b + l * DIM, xf, xb);
  }

  addr_kernel<<<1, 1024, 0, stream>>>(addr, poi, atype, Wa, ba, aa);
  final_kernel<<<TOK / 8, 256, 0, stream>>>(xf, Wout, aa, Wcomb, oo);
  lsm_kernel<<<BATCH, 256, 0, stream>>>(oo, dlen, out);
}
